// EncoderLayer_42125039239588
// MI455X (gfx1250) — matrix operand order repaired and checked by reading, not compiled, not run on hardware
//
#include <hip/hip_runtime.h>
#include <hip/hip_bf16.h>
#include <stdint.h>

// ---------------- problem constants ----------------
static constexpr int kB   = 8;
static constexpr int kS   = 1024;
static constexpr int kD   = 256;
static constexpr int kDQ  = 64;
static constexpr int kH   = 7;
static constexpr int kDFF = 256;
static constexpr float kSCALE = 0.125f;     // 1/sqrt(64)
static constexpr float kEPS   = 1e-5f;

typedef _Float16 half16 __attribute__((ext_vector_type(16)));
typedef float    float8 __attribute__((ext_vector_type(8)));

__device__ __forceinline__ float8 Z8() {
  float8 z = {0.f,0.f,0.f,0.f,0.f,0.f,0.f,0.f};
  return z;
}

__device__ __forceinline__ float8 wmma(half16 a, half16 b, float8 c) {
  return __builtin_amdgcn_wmma_f32_16x16x32_f16(false, a, false, b, (short)0, c, false, false);
}

// A fragment (16x32, M x K) from row-major matrix, stride ld (halves).
// Per-lane data is one contiguous 32B run (k = 16*half + i) -> vectorizes to b128 loads.
__device__ __forceinline__ half16 load_a(const _Float16* p, int ld, int lane) {
  int row = lane & 15;
  int kh  = (lane >> 4) * 16;
  const _Float16* q = p + (long)row * ld + kh;
  half16 a;
#pragma unroll
  for (int i = 0; i < 8; ++i) { a[i] = q[i]; a[8 + i] = q[8 + i]; }
  return a;
}

// B fragment of M^T where M is row-major [16 x K] stride ld -> one contiguous
// 32B run per lane.
__device__ __forceinline__ half16 load_bt(const _Float16* p, int ld, int lane) {
  int col = lane & 15;
  int kb  = (lane >> 4) * 16;
  const _Float16* q = p + (long)col * ld + kb;
  half16 b;
#pragma unroll
  for (int i = 0; i < 16; ++i) b[i] = q[i];
  return b;
}

// B fragment from pre-packed fragment-order weights: frag fr, lane's 16 halves
// contiguous (32B) -> two global_load_b128.
__device__ __forceinline__ half16 load_pb(const _Float16* packed, int fr, int lane) {
  return *(const half16*)(packed + ((long)fr * 32 + lane) * 16);
}

// ---------------- K0a: f32 -> f16 convert (x only) ----------------
__global__ void cvt_f16_kernel(const float* __restrict__ in, _Float16* __restrict__ out, int n) {
  int i = blockIdx.x * blockDim.x + threadIdx.x;
  if (i < n) out[i] = (_Float16)in[i];
}

// ---------------- K0b: pack f32 row-major [nmat][K][N] into B-fragment order ----
// out layout: [nmat][F][32 lanes][16 halves], F = (K/32)*(N/16), fr = kt*(N/16)+nt
__global__ void pack_b_kernel(const float* __restrict__ in, _Float16* __restrict__ out,
                              int K, int N) {
  int lane = threadIdx.x & 31;
  int fr   = blockIdx.x * (blockDim.x >> 5) + (threadIdx.x >> 5);
  int mat  = blockIdx.y;
  int Nt   = N / 16;
  int F    = (K / 32) * Nt;
  if (fr >= F) return;
  int kt = fr / Nt, nt = fr % Nt;
  int col = lane & 15, kb = (lane >> 4) * 16;
  const float* src = in + (long)mat * K * N;
  _Float16* dst = out + ((long)mat * F + fr) * 32 * 16 + (long)lane * 16;
#pragma unroll
  for (int i = 0; i < 16; ++i)
    dst[i] = (_Float16)src[(long)(kt * 32 + kb + i) * N + nt * 16 + col];
}

// ---------------- K1: QKV projection ----------------
// grid (B*S/16/8, H), block 256 (8 waves). One wave -> one 16-row tile.
// Q,K stored row-major [s][q]; V stored TRANSPOSED [v][s].
__global__ void qkv_kernel(const _Float16* __restrict__ xh,
                           const _Float16* __restrict__ wqp,
                           const _Float16* __restrict__ wkp,
                           const _Float16* __restrict__ wvp,
                           _Float16* __restrict__ Qo,
                           _Float16* __restrict__ Ko,
                           _Float16* __restrict__ Vt) {
  int lane = threadIdx.x & 31;
  int wave = threadIdx.x >> 5;
  int h    = blockIdx.y;
  long tile = (long)blockIdx.x * 8 + wave;           // over B*S/16
  long g0   = tile * 16;
  int  b    = (int)(g0 / kS);
  int  s0   = (int)(g0 % kS);
  int  bh   = b * kH + h;

  const _Float16* xrow = xh + g0 * kD;
  half16 a[8];
#pragma unroll
  for (int k = 0; k < 8; ++k) a[k] = load_a(xrow + k * 32, kD, lane);

  const long wstride = (long)kD * kDQ;               // elements per matrix
  const _Float16* Wp[3] = { wqp + (long)h * wstride, wkp + (long)h * wstride,
                            wvp + (long)h * wstride };
  int col = lane & 15, hf = lane >> 4;

  _Float16* Qb = Qo + ((long)bh * kS + s0) * kDQ;
  _Float16* Kb = Ko + ((long)bh * kS + s0) * kDQ;
  _Float16* Vb = Vt + (long)bh * kDQ * kS;           // [64][1024]

#pragma unroll
  for (int m = 0; m < 3; ++m) {
#pragma unroll
    for (int nt = 0; nt < 4; ++nt) {
      // issue all 8 fragment loads up-front, then the wmma chain
      half16 wb[8];
#pragma unroll
      for (int k = 0; k < 8; ++k) wb[k] = load_pb(Wp[m], k * 4 + nt, lane);
      float8 c = Z8();
#pragma unroll
      for (int k = 0; k < 8; ++k) c = wmma(a[k], wb[k], c);

      _Float16 t[8];
#pragma unroll
      for (int r = 0; r < 8; ++r) t[r] = (_Float16)c[r];
      if (m < 2) {
        _Float16* o = (m == 0 ? Qb : Kb) + nt * 16 + col;
#pragma unroll
        for (int r = 0; r < 8; ++r) o[(long)(8 * hf + r) * kDQ] = t[r];
      } else {
        // transposed store: V^T[v][s], 8 contiguous halves per lane
        _Float16* o = Vb + (long)(nt * 16 + col) * kS + s0 + 8 * hf;
#pragma unroll
        for (int r = 0; r < 8; ++r) o[r] = t[r];
      }
    }
  }
}

// ---------------- K2: flash attention (transposed scores) ----------------
// grid (B*H, S/16/8), block 256. One wave -> one 16-query tile, streams all keys.
__global__ void attn_kernel(const _Float16* __restrict__ Qh,
                            const _Float16* __restrict__ Kh,
                            const _Float16* __restrict__ Vt,
                            const uint8_t* __restrict__ gmask,
                            const uint8_t* __restrict__ lmask,
                            _Float16* __restrict__ ctxh) {
  int lane = threadIdx.x & 31;
  int wave = threadIdx.x >> 5;
  int bh = blockIdx.x;
  int b = bh / kH, h = bh % kH;
  int m0 = (blockIdx.y * 8 + wave) * 16;

  const _Float16* Q  = Qh + (long)bh * kS * kDQ;
  const _Float16* K  = Kh + (long)bh * kS * kDQ;
  const _Float16* VT = Vt + (long)bh * kDQ * kS;     // [64][1024]
  const uint8_t* mask = (h == 0) ? (lmask + (long)b * kS * kS)
                                 : (gmask + ((long)b * 6 + (h - 1)) * (long)kS * kS);

  int col = lane & 15, hf = lane >> 4;
  // Q^T B-fragments (contiguous loads), reused for all key tiles
  half16 bq0 = load_bt(Q + (long)m0 * kDQ, kDQ, lane);
  half16 bq1 = load_bt(Q + (long)m0 * kDQ + 32, kDQ, lane);

  const uint8_t* mrow = mask + (long)(m0 + col) * kS + 8 * hf;

  float runM = -1e30f, runL = 0.f;
  float8 acc[4] = { Z8(), Z8(), Z8(), Z8() };

  for (int t0 = 0; t0 < kS; t0 += 32) {
    // ---- issue ALL independent loads for this iteration up-front ----
    half16 ak0 = load_a(K + (long)t0 * kDQ, kDQ, lane);
    half16 ak1 = load_a(K + (long)t0 * kDQ + 32, kDQ, lane);
    half16 ak2 = load_a(K + (long)(t0 + 16) * kDQ, kDQ, lane);
    half16 ak3 = load_a(K + (long)(t0 + 16) * kDQ + 32, kDQ, lane);
    unsigned long long mb0 = *(const unsigned long long*)(mrow + t0);
    unsigned long long mb1 = *(const unsigned long long*)(mrow + t0 + 16);
    half16 va[4];
#pragma unroll
    for (int vt = 0; vt < 4; ++vt)
      va[vt] = load_a(VT + (long)(vt * 16) * kS + t0, kS, lane);
    // prefetch mask stream 8 iterations ahead (speculative; OOB dropped by HW)
    __builtin_prefetch(mrow + t0 + 256);

    // ---- scores (transposed): S^T = K . Q^T ----
    float8 c0 = wmma(ak0, bq0, Z8());
    c0 = wmma(ak1, bq1, c0);
    float8 c1 = wmma(ak2, bq0, Z8());
    c1 = wmma(ak3, bq1, c1);

    float p[2][8];
    unsigned mk[2] = {0u, 0u};
    float tmax = -1e30f;
#pragma unroll
    for (int r = 0; r < 8; ++r) {
      float sv0, sv1;
      if ((mb0 >> (8 * r)) & 0xffULL) { mk[0] |= (1u << r); sv0 = -1e30f; }
      else sv0 = c0[r] * kSCALE;
      if ((mb1 >> (8 * r)) & 0xffULL) { mk[1] |= (1u << r); sv1 = -1e30f; }
      else sv1 = c1[r] * kSCALE;
      p[0][r] = sv0; p[1][r] = sv1;
      tmax = fmaxf(tmax, fmaxf(sv0, sv1));
    }
    // combine max across the lane pair sharing this query column
    tmax = fmaxf(tmax, __shfl_xor(tmax, 16));
    float newM  = fmaxf(runM, tmax);
    float alpha = __expf(runM - newM);
    runM = newM;

    float lsum = 0.f;
#pragma unroll
    for (int r = 0; r < 8; ++r) {
      p[0][r] = (mk[0] >> r & 1u) ? 0.f : __expf(p[0][r] - newM);
      p[1][r] = (mk[1] >> r & 1u) ? 0.f : __expf(p[1][r] - newM);
      lsum += p[0][r] + p[1][r];
    }
    runL = runL * alpha + lsum;
#pragma unroll
    for (int vt = 0; vt < 4; ++vt) acc[vt] = acc[vt] * alpha;

    // P^T tile (C layout) -> B fragment via shfl_xor(16): half0 needs rows 0..15
    // (tile0), half1 needs rows 16..31 (tile1); each half owns 8 of them.
    half16 bp;
#pragma unroll
    for (int r = 0; r < 8; ++r) {
      float x0 = __shfl_xor(p[0][r], 16);
      float x1 = __shfl_xor(p[1][r], 16);
      bp[r]     = (_Float16)(hf ? x1 : p[0][r]);
      bp[8 + r] = (_Float16)(hf ? p[1][r] : x0);
    }
    // context^T += V^T @ P^T (V frags already in registers)
#pragma unroll
    for (int vt = 0; vt < 4; ++vt)
      acc[vt] = wmma(va[vt], bp, acc[vt]);
  }

  float lt  = runL + __shfl_xor(runL, 16);
  float inv = lt > 0.f ? 1.f / lt : 0.f;

  // store context row-major [s][v] as f16 (8 contiguous halves per lane per tile)
  _Float16* cb = ctxh + (long)bh * kS * kDQ + (long)(m0 + col) * kDQ;
#pragma unroll
  for (int vt = 0; vt < 4; ++vt)
#pragma unroll
    for (int r = 0; r < 8; ++r)
      cb[16 * vt + 8 * hf + r] = (_Float16)(acc[vt][r] * inv);
}

// ---------------- K3: per-branch Wo proj + residual + LayerNorm + sum ----------------
// grid B*S/16/8, block 256. One wave -> one 16-row tile, loops over 7 branches.
__global__ void branch_kernel(const _Float16* __restrict__ ctxh,
                              const _Float16* __restrict__ wop,
                              const float* __restrict__ x,
                              float* __restrict__ summed,
                              _Float16* __restrict__ sumh) {
  int lane = threadIdx.x & 31;
  int wave = threadIdx.x >> 5;
  long tile = (long)blockIdx.x * 8 + wave;
  long g0   = tile * 16;
  int b = (int)(g0 / kS), s0 = (int)(g0 % kS);
  int col = lane & 15, hf = lane >> 4;

  float sum[16][8];
#pragma unroll
  for (int nt = 0; nt < 16; ++nt)
#pragma unroll
    for (int r = 0; r < 8; ++r) sum[nt][r] = 0.f;

  for (int h = 0; h < kH; ++h) {
    const _Float16* cb = ctxh + ((long)(b * kH + h) * kS + s0) * kDQ;
    half16 a0 = load_a(cb, kDQ, lane);
    half16 a1 = load_a(cb + 32, kDQ, lane);
    const _Float16* wo = wop + (long)h * kDQ * kD;   // packed: F=32, Nt=16

    float o[16][8];
#pragma unroll
    for (int nt = 0; nt < 16; ++nt) {
      half16 b0 = load_pb(wo, nt, lane);
      half16 b1 = load_pb(wo, 16 + nt, lane);
      float8 c = wmma(a0, b0, Z8());
      c = wmma(a1, b1, c);
#pragma unroll
      for (int r = 0; r < 8; ++r)
        o[nt][r] = c[r] + x[(g0 + 8 * hf + r) * kD + nt * 16 + col];
    }
    // LayerNorm per row (row = 8*hf + r, shared by 16 lanes of the half)
#pragma unroll
    for (int r = 0; r < 8; ++r) {
      float s1 = 0.f, s2 = 0.f;
#pragma unroll
      for (int nt = 0; nt < 16; ++nt) { float v = o[nt][r]; s1 += v; s2 += v * v; }
#pragma unroll
      for (int m = 1; m < 16; m <<= 1) { s1 += __shfl_xor(s1, m); s2 += __shfl_xor(s2, m); }
      float mu  = s1 * (1.f / 256.f);
      float var = s2 * (1.f / 256.f) - mu * mu;
      float rs  = rsqrtf(var + kEPS);
#pragma unroll
      for (int nt = 0; nt < 16; ++nt) sum[nt][r] += (o[nt][r] - mu) * rs;
    }
  }
#pragma unroll
  for (int nt = 0; nt < 16; ++nt)
#pragma unroll
    for (int r = 0; r < 8; ++r) {
      long idx = (g0 + 8 * hf + r) * kD + nt * 16 + col;
      summed[idx] = sum[nt][r];
      sumh[idx]   = (_Float16)sum[nt][r];
    }
}

// ---------------- K4: FFN (W1,relu,W2) + residual + final LayerNorm ----------------
// grid B*S/16/4, block 128 (4 waves), 32KB LDS for the relu-tile transpose bounce.
__global__ __launch_bounds__(128)
void ffn_kernel(const _Float16* __restrict__ sh,
                const float* __restrict__ sf,
                const _Float16* __restrict__ w1p,
                const _Float16* __restrict__ w2p,
                float* __restrict__ out) {
  __shared__ _Float16 lds[4 * 16 * kDFF];
  int lane = threadIdx.x & 31;
  int wave = threadIdx.x >> 5;
  long tile = (long)blockIdx.x * 4 + wave;
  long g0   = tile * 16;
  int col = lane & 15, hf = lane >> 4;
  _Float16* lbuf = lds + wave * 16 * kDFF;

  half16 a[8];
#pragma unroll
  for (int k = 0; k < 8; ++k) a[k] = load_a(sh + g0 * kD + k * 32, kD, lane);

  // hidden = relu(summed @ W1), written to LDS in row-major f16
#pragma unroll
  for (int nt = 0; nt < 16; ++nt) {
    half16 wb[8];
#pragma unroll
    for (int k = 0; k < 8; ++k) wb[k] = load_pb(w1p, k * 16 + nt, lane);
    float8 c = Z8();
#pragma unroll
    for (int k = 0; k < 8; ++k) c = wmma(a[k], wb[k], c);
#pragma unroll
    for (int r = 0; r < 8; ++r)
      lbuf[(8 * hf + r) * kDFF + nt * 16 + col] = (_Float16)fmaxf(c[r], 0.f);
  }
  __syncthreads();

  half16 a2[8];
#pragma unroll
  for (int k = 0; k < 8; ++k) a2[k] = load_a(lbuf + k * 32, kDFF, lane);

  float o[16][8];
#pragma unroll
  for (int nt = 0; nt < 16; ++nt) {
    half16 wb[8];
#pragma unroll
    for (int k = 0; k < 8; ++k) wb[k] = load_pb(w2p, k * 16 + nt, lane);
    float8 c = Z8();
#pragma unroll
    for (int k = 0; k < 8; ++k) c = wmma(a2[k], wb[k], c);
#pragma unroll
    for (int r = 0; r < 8; ++r)
      o[nt][r] = c[r] + sf[(g0 + 8 * hf + r) * kD + nt * 16 + col];
  }
  // final LayerNorm
#pragma unroll
  for (int r = 0; r < 8; ++r) {
    float s1 = 0.f, s2 = 0.f;
#pragma unroll
    for (int nt = 0; nt < 16; ++nt) { float v = o[nt][r]; s1 += v; s2 += v * v; }
#pragma unroll
    for (int m = 1; m < 16; m <<= 1) { s1 += __shfl_xor(s1, m); s2 += __shfl_xor(s2, m); }
    float mu  = s1 * (1.f / 256.f);
    float var = s2 * (1.f / 256.f) - mu * mu;
    float rs  = rsqrtf(var + kEPS);
#pragma unroll
    for (int nt = 0; nt < 16; ++nt)
      out[(g0 + 8 * hf + r) * kD + nt * 16 + col] = (o[nt][r] - mu) * rs;
  }
}

// ---------------- host launch ----------------
extern "C" void kernel_launch(void* const* d_in, const int* in_sizes, int n_in,
                              void* d_out, int out_size, void* d_ws, size_t ws_size,
                              hipStream_t stream) {
  (void)in_sizes; (void)n_in; (void)out_size; (void)ws_size;
  const float*   x     = (const float*)d_in[0];
  const uint8_t* gmask = (const uint8_t*)d_in[1];   // [B,6,S,S] bool
  const uint8_t* lmask = (const uint8_t*)d_in[2];   // [B,S,S]   bool
  const float*   Wq = (const float*)d_in[3];
  const float*   Wk = (const float*)d_in[4];
  const float*   Wv = (const float*)d_in[5];
  const float*   Wo = (const float*)d_in[6];
  const float*   W1 = (const float*)d_in[7];
  const float*   W2 = (const float*)d_in[8];
  float* out = (float*)d_out;

  // workspace carve-up (256B aligned)
  uint8_t* ws = (uint8_t*)d_ws;
  size_t off = 0;
  auto take = [&](size_t bytes) -> void* {
    void* p = ws + off;
    off = (off + bytes + 255) & ~(size_t)255;
    return p;
  };
  const size_t nX   = (size_t)kB * kS * kD;          // 2M
  const size_t nW   = (size_t)kH * kD * kDQ;         // 114688 (same count packed)
  const size_t nFF  = (size_t)kD * kDFF;             // 65536
  const size_t nQKV = (size_t)kB * kH * kS * kDQ;    // 3.67M

  _Float16* xh  = (_Float16*)take(nX * 2);
  _Float16* wqp = (_Float16*)take(nW * 2);
  _Float16* wkp = (_Float16*)take(nW * 2);
  _Float16* wvp = (_Float16*)take(nW * 2);
  _Float16* wop = (_Float16*)take(nW * 2);
  _Float16* w1p = (_Float16*)take(nFF * 2);
  _Float16* w2p = (_Float16*)take(nFF * 2);
  _Float16* Qh  = (_Float16*)take(nQKV * 2);
  _Float16* Kh  = (_Float16*)take(nQKV * 2);
  _Float16* Vt  = (_Float16*)take(nQKV * 2);
  _Float16* ctx = (_Float16*)take(nQKV * 2);
  float*    smf = (float*)take(nX * 4);
  _Float16* smh = (_Float16*)take(nX * 2);

  cvt_f16_kernel<<<dim3((unsigned)((nX + 255) / 256)), 256, 0, stream>>>(x, xh, (int)nX);

  auto pack = [&](const float* in, _Float16* o, int K, int N, int nmat) {
    int F = (K / 32) * (N / 16);
    pack_b_kernel<<<dim3((F + 7) / 8, nmat), 256, 0, stream>>>(in, o, K, N);
  };
  pack(Wq, wqp, kD, kDQ, kH);    // F=32 per h
  pack(Wk, wkp, kD, kDQ, kH);
  pack(Wv, wvp, kD, kDQ, kH);
  pack(Wo, wop, kDQ, kD, kH);    // F=32 per h (Kt=2, Nt=16)
  pack(W1, w1p, kD, kDFF, 1);    // F=128
  pack(W2, w2p, kDFF, kD, 1);    // F=128

  // QKV: (B*S/16)/8 x H blocks of 8 waves
  qkv_kernel<<<dim3((kB * kS / 16) / 8, kH), 256, 0, stream>>>(xh, wqp, wkp, wvp, Qh, Kh, Vt);

  // attention: (B*H) x (S/16/8) blocks of 8 waves
  attn_kernel<<<dim3(kB * kH, (kS / 16) / 8), 256, 0, stream>>>(Qh, Kh, Vt, gmask, lmask, ctx);

  // branch combine
  branch_kernel<<<dim3((kB * kS / 16) / 8), 256, 0, stream>>>(ctx, wop, x, smf, smh);

  // FFN + final LN
  ffn_kernel<<<dim3((kB * kS / 16) / 4), 128, 0, stream>>>(smh, smf, w1p, w2p, out);
}
